// Encoder_79096117723504
// MI455X (gfx1250) — compile-verified, matrix operand order checked by reading, not run on hardware
//
#include <hip/hip_runtime.h>

// ---------------------------------------------------------------------------
// CollaborativeAttention encoder for MI455X (gfx1250, wave32, WMMA bf16).
// Dominant cost: per-head full-D QK^T (16 x [2048x768]x[768x2048] per
// instance) -> v_wmma_f32_16x16x32_bf16, mix folded into Q at staging time.
// Flash kernel uses 32 query rows per wave so every K/V fragment loaded from
// L2 feeds two WMMAs (halves L2 traffic vs 16-row tiles).
// ---------------------------------------------------------------------------

typedef __bf16 bf16;
typedef __attribute__((ext_vector_type(16))) __bf16 v16bf;
typedef __attribute__((ext_vector_type(8)))  __bf16 v8bf;
typedef __attribute__((ext_vector_type(4)))  __bf16 v4bf;
typedef __attribute__((ext_vector_type(8)))  float  v8f;
typedef __attribute__((ext_vector_type(4)))  float  v4f;

constexpr int kD   = 768;
constexpr int kF   = 2048;
constexpr int kH   = 16;
constexpr int kDH  = 48;
constexpr size_t kMat = (size_t)kD * kD;     // 589824
constexpr size_t kFD  = (size_t)kF * kD;     // 1572864

// workspace layout (bytes)
constexpr size_t OFF_WBF = 0;                                 // 8 mats bf16
constexpr size_t OFF_GBF = OFF_WBF + 8 * kMat * 2;            // 6 gathered bf16
constexpr size_t OFF_KBF = OFF_GBF + 6 * kFD * 2;             // K bf16 per inst
constexpr size_t OFF_VT  = OFF_KBF + 6 * kFD * 2;             // V^T bf16 per inst
constexpr size_t OFF_Q   = OFF_VT  + 6 * kFD * 2;             // Q f32 (later LN buf)
constexpr size_t OFF_CB  = OFF_Q   + 6 * kFD * 4;             // cb f32 [inst][16][2048]
constexpr size_t OFF_CTX = OFF_CB  + 6 * (size_t)kH * kF * 4; // ctx f32 per inst

// ---------------------------------------------------------------------------
// Fragment helpers. 16x32 bf16 A/B fragment from a row-major matrix with
// leading dim ld: lane L holds row (L&15), K-chunks at (L>>4)*8 and +16.
// ---------------------------------------------------------------------------
__device__ inline v16bf ldfrag(const bf16* base, int ld, int lane) {
  const bf16* p = base + (size_t)(lane & 15) * ld + ((lane >> 4) << 3);
  v8bf lo = *reinterpret_cast<const v8bf*>(p);
  v8bf hi = *reinterpret_cast<const v8bf*>(p + 16);
  return __builtin_shufflevector(lo, hi, 0,1,2,3,4,5,6,7,8,9,10,11,12,13,14,15);
}

__device__ inline v16bf ldfrag_f32(const float* base, int ld, int lane) {
  const float* p = base + (size_t)(lane & 15) * ld + ((lane >> 4) << 3);
  v4f a0 = *reinterpret_cast<const v4f*>(p);
  v4f a1 = *reinterpret_cast<const v4f*>(p + 4);
  v4f a2 = *reinterpret_cast<const v4f*>(p + 16);
  v4f a3 = *reinterpret_cast<const v4f*>(p + 20);
  v16bf a;
#pragma unroll
  for (int i = 0; i < 4; ++i) {
    a[i]      = (bf16)a0[i];
    a[4 + i]  = (bf16)a1[i];
    a[8 + i]  = (bf16)a2[i];
    a[12 + i] = (bf16)a3[i];
  }
  return a;
}

__device__ inline v8f wmma_bf16(v16bf a, v16bf b, v8f c) {
  return __builtin_amdgcn_wmma_f32_16x16x32_bf16(false, a, false, b,
                                                 (short)0, c, false, false);
}

// ---------------------------------------------------------------------------
// Weight f32 -> bf16, layout wbf[set*4+mat][768*768], mat: 0=Wq 1=Wk 2=Wv 3=Wd
// ---------------------------------------------------------------------------
__global__ __launch_bounds__(256) void k_wcvt(
    const float* wq0, const float* wk0, const float* wv0, const float* wd0,
    const float* wq1, const float* wk1, const float* wv1, const float* wd1,
    bf16* __restrict__ wbf) {
  size_t i = (size_t)blockIdx.x * 256 + threadIdx.x;
  if (i >= 8 * kMat) return;
  int mat = (int)(i / kMat);
  size_t off = i - (size_t)mat * kMat;
  const float* s;
  switch (mat) {
    case 0: s = wq0; break; case 1: s = wk0; break;
    case 2: s = wv0; break; case 3: s = wd0; break;
    case 4: s = wq1; break; case 5: s = wk1; break;
    case 6: s = wv1; break; default: s = wd1; break;
  }
  wbf[i] = (bf16)s[off];
}

// gathered activations, bf16: g[b][side][2048][768], side0=h[fpos], side1=h[tpos]
__global__ __launch_bounds__(256) void k_gather(
    const float* __restrict__ h, const int* __restrict__ fpos,
    const int* __restrict__ tpos, bf16* __restrict__ g) {
  size_t i = (size_t)blockIdx.x * 256 + threadIdx.x;
  if (i >= 6 * kFD) return;
  int d  = (int)(i % kD);
  int t  = (int)((i / kD) % kF);
  int bs = (int)(i / kFD);
  int b = bs >> 1, side = bs & 1;
  const int* idx = (side ? tpos : fpos) + b * kF;
  g[i] = (bf16)h[((size_t)b * 512 + (idx[t] & 511)) * kD + d];
}

// content bias: cb[inst][hh][t] = dot(h[b][to[t]], Wcb[hh])
__global__ __launch_bounds__(256) void k_cb(
    const float* __restrict__ h, const int* __restrict__ fpos,
    const int* __restrict__ tpos, const float* __restrict__ wcb_q,
    const float* __restrict__ wcb_c, float* __restrict__ cb) {
  int gid = blockIdx.x * 256 + threadIdx.x;
  if (gid >= 6 * kH * kF) return;
  int t = gid % kF;
  int hh = (gid / kF) % kH;
  int inst = gid / (kF * kH);
  int b = inst >> 1, dir = inst & 1;
  const int* tidx = (dir ? fpos : tpos) + b * kF;       // "to" side
  const float* wr = (dir ? wcb_c : wcb_q) + hh * kD;
  const float* hr = h + ((size_t)b * 512 + (tidx[t] & 511)) * kD;
  float s = 0.f;
  for (int d = 0; d < kD; ++d) s += hr[d] * wr[d];
  cb[((size_t)inst * kH + hh) * kF + t] = s;
}

// ---------------------------------------------------------------------------
// Projection GEMM: C[2048x768] = A_bf16 @ W_bf16^T. One wave per 16x64 tile.
// MODE 0: store f32; MODE 1: store bf16; MODE 2: store bf16 transposed + bias.
// ---------------------------------------------------------------------------
template <int MODE>
__global__ __launch_bounds__(32) void k_proj_gemm(
    const bf16* __restrict__ A, const bf16* __restrict__ W,
    float* __restrict__ Cf, bf16* __restrict__ Cb,
    const float* __restrict__ bias) {
  int lane = threadIdx.x;
  int m0 = blockIdx.x * 16;
  int n0 = blockIdx.y * 64;
  v8f acc[4] = {};
  for (int kk = 0; kk < kD; kk += 32) {
    v16bf a = ldfrag(A + (size_t)m0 * kD + kk, kD, lane);
#pragma unroll
    for (int j = 0; j < 4; ++j) {
      v16bf bfr = ldfrag(W + (size_t)(n0 + j * 16) * kD + kk, kD, lane);
      acc[j] = wmma_bf16(a, bfr, acc[j]);
    }
  }
  int mr8 = (lane >> 4) << 3;
  int nc = lane & 15;
#pragma unroll
  for (int j = 0; j < 4; ++j) {
    int n = n0 + j * 16 + nc;
    float bv = (MODE == 2) ? bias[n] : 0.f;
#pragma unroll
    for (int r = 0; r < 8; ++r) {
      int m = m0 + r + mr8;
      float v = acc[j][r] + bv;
      if (MODE == 0)      Cf[(size_t)m * kD + n] = v;
      else if (MODE == 1) Cb[(size_t)m * kD + n] = (bf16)v;
      else                Cb[(size_t)n * kF + m] = (bf16)v;   // V^T
    }
  }
}

// ---------------------------------------------------------------------------
// Fused flash attention. One wave per block; wave owns 32 query rows (two
// 16-row M-tiles) of one head of one instance, so every K/V fragment loaded
// from L2 feeds two WMMAs. Q*mix staged once in LDS as bf16 (48KB); P is
// transposed through a 2KB LDS buffer between the two WMMA stages (DS is
// in-order per wave; s_wait_dscnt 0 drains before the A-frag reload).
// ---------------------------------------------------------------------------
__global__ __launch_bounds__(32) void k_flash(
    const float* __restrict__ qf, const bf16* __restrict__ kbf,
    const bf16* __restrict__ vT, const float* __restrict__ cb,
    const float* __restrict__ mix_q, const float* __restrict__ mix_c,
    float* __restrict__ ctxout) {
  int inst = blockIdx.z;
  int dir = inst & 1;
  int hh = blockIdx.y;
  int lane = threadIdx.x;
  int f0 = blockIdx.x * 32;

  const float* q  = qf  + (size_t)inst * kFD;
  const bf16*  K  = kbf + (size_t)inst * kFD;
  const bf16*  V  = vT  + (size_t)inst * kFD + (size_t)(hh * kDH) * kF;
  const float* CB = cb  + ((size_t)inst * kH + hh) * kF;
  const float* mix = (dir ? mix_c : mix_q) + hh * kD;

  __shared__ __align__(16) bf16 qs[32 * kD];   // 48 KB
  __shared__ __align__(16) bf16 ps[32 * 32];   // 2 KB

  // stage Q * mix[h] as bf16 (fold per-head mixing into Q); vectorized x4
  for (int i = lane * 4; i < 32 * kD; i += 32 * 4) {
    int r = i / kD, d = i - r * kD;
    v4f x = *reinterpret_cast<const v4f*>(&q[(size_t)(f0 + r) * kD + d]);
    v4f mx = *reinterpret_cast<const v4f*>(&mix[d]);
    v4bf y;
#pragma unroll
    for (int e = 0; e < 4; ++e) y[e] = (bf16)(x[e] * mx[e]);
    *reinterpret_cast<v4bf*>(&qs[i]) = y;
  }
  asm volatile("s_wait_dscnt 0" ::: "memory");

  v8f ctx[2][3] = {};
  float Mr[2][8], Lr[2][8];
#pragma unroll
  for (int u = 0; u < 2; ++u)
#pragma unroll
    for (int r = 0; r < 8; ++r) { Mr[u][r] = -1e30f; Lr[u][r] = 0.f; }

  const float iscale = 0.14433756729740643f;  // 1/sqrt(768/16)
  int nc = lane & 15;
  int mr8 = (lane >> 4) << 3;

  for (int t0 = 0; t0 < kF; t0 += 32) {
    v8f sAcc[2][2] = {};
    for (int kk = 0; kk < kD; kk += 32) {
      v16bf b0 = ldfrag(K + (size_t)t0 * kD + kk, kD, lane);
      v16bf b1 = ldfrag(K + (size_t)(t0 + 16) * kD + kk, kD, lane);
#pragma unroll
      for (int u = 0; u < 2; ++u) {
        v16bf a = ldfrag(&qs[(size_t)(u * 16) * kD + kk], kD, lane);
        sAcc[u][0] = wmma_bf16(a, b0, sAcc[u][0]);
        sAcc[u][1] = wmma_bf16(a, b1, sAcc[u][1]);
      }
    }
    float c0 = CB[t0 + nc], c1 = CB[t0 + 16 + nc];
#pragma unroll
    for (int u = 0; u < 2; ++u) {
#pragma unroll
      for (int r = 0; r < 8; ++r) {
        float x0 = (sAcc[u][0][r] + c0) * iscale;
        float x1 = (sAcc[u][1][r] + c1) * iscale;
        float mt = fmaxf(x0, x1);                 // row reduce in 16-lane half
        mt = fmaxf(mt, __shfl_xor(mt, 1));
        mt = fmaxf(mt, __shfl_xor(mt, 2));
        mt = fmaxf(mt, __shfl_xor(mt, 4));
        mt = fmaxf(mt, __shfl_xor(mt, 8));
        float Mn = fmaxf(Mr[u][r], mt);
        float alpha = __expf(Mr[u][r] - Mn);
        Mr[u][r] = Mn;
        float p0 = __expf(x0 - Mn);
        float p1 = __expf(x1 - Mn);
        int m = u * 16 + r + mr8;
        ps[m * 32 + nc]      = (bf16)p0;          // C-frag -> A-layout transpose
        ps[m * 32 + nc + 16] = (bf16)p1;
        float rs = p0 + p1;
        rs += __shfl_xor(rs, 1);
        rs += __shfl_xor(rs, 2);
        rs += __shfl_xor(rs, 4);
        rs += __shfl_xor(rs, 8);
        Lr[u][r] = Lr[u][r] * alpha + rs;
        ctx[u][0][r] *= alpha; ctx[u][1][r] *= alpha; ctx[u][2][r] *= alpha;
      }
    }
    asm volatile("s_wait_dscnt 0" ::: "memory");  // drain P stores (DS in-order)
    v16bf vb0 = ldfrag(V + 0 * 16 * (size_t)kF + t0, kF, lane);
    v16bf vb1 = ldfrag(V + 1 * 16 * (size_t)kF + t0, kF, lane);
    v16bf vb2 = ldfrag(V + 2 * 16 * (size_t)kF + t0, kF, lane);
#pragma unroll
    for (int u = 0; u < 2; ++u) {
      v16bf pf = ldfrag(&ps[u * 16 * 32], 32, lane);
      ctx[u][0] = wmma_bf16(pf, vb0, ctx[u][0]);
      ctx[u][1] = wmma_bf16(pf, vb1, ctx[u][1]);
      ctx[u][2] = wmma_bf16(pf, vb2, ctx[u][2]);
    }
  }

  float* co = ctxout + (size_t)inst * kFD;
#pragma unroll
  for (int u = 0; u < 2; ++u) {
#pragma unroll
    for (int r = 0; r < 8; ++r) {
      float inv = 1.0f / Lr[u][r];
      size_t row = (size_t)(f0 + u * 16 + r + mr8) * kD + hh * kDH;
      co[row + 0 * 16 + nc] = ctx[u][0][r] * inv;
      co[row + 1 * 16 + nc] = ctx[u][1][r] * inv;
      co[row + 2 * 16 + nc] = ctx[u][2][r] * inv;
    }
  }
}

// ---------------------------------------------------------------------------
// Output projection + bias + residual (re-gather): lnin = ctx@Wd^T + bd + fs
// ---------------------------------------------------------------------------
__global__ __launch_bounds__(32) void k_out_proj(
    const float* __restrict__ ctx, const bf16* __restrict__ wbf,
    const float* __restrict__ bd_q, const float* __restrict__ bd_c,
    const float* __restrict__ h, const int* __restrict__ fpos,
    const int* __restrict__ tpos, float* __restrict__ lnin) {
  int inst = blockIdx.z;
  int b = inst >> 1, dir = inst & 1;
  const float* A  = ctx + (size_t)inst * kFD;
  const bf16*  Wd = wbf + (size_t)(dir * 4 + 3) * kMat;
  const float* bd = dir ? bd_c : bd_q;
  const int* fidx = (dir ? tpos : fpos) + b * kF;
  const float* hb = h + (size_t)b * 512 * kD;

  int lane = threadIdx.x;
  int m0 = blockIdx.x * 16;
  int n0 = blockIdx.y * 64;
  v8f acc[4] = {};
  for (int kk = 0; kk < kD; kk += 32) {
    v16bf a = ldfrag_f32(A + (size_t)m0 * kD + kk, kD, lane);
#pragma unroll
    for (int j = 0; j < 4; ++j) {
      v16bf bfr = ldfrag(Wd + (size_t)(n0 + j * 16) * kD + kk, kD, lane);
      acc[j] = wmma_bf16(a, bfr, acc[j]);
    }
  }
  int mr8 = (lane >> 4) << 3;
  int nc = lane & 15;
  float* out = lnin + (size_t)inst * kFD;
#pragma unroll
  for (int r = 0; r < 8; ++r) {
    int m = m0 + r + mr8;
    int src = fidx[m] & 511;
#pragma unroll
    for (int j = 0; j < 4; ++j) {
      int n = n0 + j * 16 + nc;
      out[(size_t)m * kD + n] = acc[j][r] + bd[n] + hb[(size_t)src * kD + n];
    }
  }
}

// LayerNorm in place on lnin rows
__global__ __launch_bounds__(256) void k_layernorm(
    float* __restrict__ lnin, const float* g_q, const float* g_c,
    const float* b_q, const float* b_c) {
  int inst = blockIdx.y;
  int dir = inst & 1;
  float* row = lnin + ((size_t)inst * kF + blockIdx.x) * kD;
  int tid = threadIdx.x;
  float x0 = row[tid], x1 = row[tid + 256], x2 = row[tid + 512];
  __shared__ float rs[256], rs2[256];
  rs[tid]  = x0 + x1 + x2;
  rs2[tid] = x0 * x0 + x1 * x1 + x2 * x2;
  __syncthreads();
  for (int o = 128; o > 0; o >>= 1) {
    if (tid < o) { rs[tid] += rs[tid + o]; rs2[tid] += rs2[tid + o]; }
    __syncthreads();
  }
  float mu = rs[0] * (1.f / kD);
  float var = rs2[0] * (1.f / kD) - mu * mu;
  float inv = rsqrtf(var + 1e-5f);
  const float* g = dir ? g_c : g_q;
  const float* bt = dir ? b_c : b_q;
  row[tid]       = (x0 - mu) * inv * g[tid]       + bt[tid];
  row[tid + 256] = (x1 - mu) * inv * g[tid + 256] + bt[tid + 256];
  row[tid + 512] = (x2 - mu) * inv * g[tid + 512] + bt[tid + 512];
}

// mean over F -> out[b][dir*768 + d]
__global__ __launch_bounds__(256) void k_mean(const float* __restrict__ lnin,
                                              float* __restrict__ out) {
  int inst = blockIdx.y;
  int d = blockIdx.x * 256 + threadIdx.x;
  const float* base = lnin + (size_t)inst * kFD + d;
  float s = 0.f;
  for (int f = 0; f < kF; ++f) s += base[(size_t)f * kD];
  int b = inst >> 1, dir = inst & 1;
  out[(size_t)b * 1536 + dir * kD + d] = s * (1.f / kF);
}

// ---------------------------------------------------------------------------
extern "C" void kernel_launch(void* const* d_in, const int* in_sizes, int n_in,
                              void* d_out, int out_size, void* d_ws,
                              size_t ws_size, hipStream_t stream) {
  (void)in_sizes; (void)n_in; (void)out_size; (void)ws_size;
  const float* h    = (const float*)d_in[0];
  const int*   fpos = (const int*)d_in[1];
  const int*   tpos = (const int*)d_in[2];
  // qtoc: 3 Wq, 4 Wk, 5 Wcb, 6 Wv, 7 Wd, 8 mix, 9 bv, 10 bd, 11 lng, 12 lnb
  // ctoq: 13..22 same order
  const float *Wq0 = (const float*)d_in[3],  *Wk0 = (const float*)d_in[4];
  const float *Wcb0 = (const float*)d_in[5], *Wv0 = (const float*)d_in[6];
  const float *Wd0 = (const float*)d_in[7],  *mix0 = (const float*)d_in[8];
  const float *bv0 = (const float*)d_in[9],  *bd0 = (const float*)d_in[10];
  const float *lng0 = (const float*)d_in[11], *lnb0 = (const float*)d_in[12];
  const float *Wq1 = (const float*)d_in[13], *Wk1 = (const float*)d_in[14];
  const float *Wcb1 = (const float*)d_in[15], *Wv1 = (const float*)d_in[16];
  const float *Wd1 = (const float*)d_in[17], *mix1 = (const float*)d_in[18];
  const float *bv1 = (const float*)d_in[19], *bd1 = (const float*)d_in[20];
  const float *lng1 = (const float*)d_in[21], *lnb1 = (const float*)d_in[22];

  char* ws = (char*)d_ws;
  bf16*  wbf = (bf16*)(ws + OFF_WBF);
  bf16*  gbf = (bf16*)(ws + OFF_GBF);
  bf16*  kbf = (bf16*)(ws + OFF_KBF);
  bf16*  vT  = (bf16*)(ws + OFF_VT);
  float* qf  = (float*)(ws + OFF_Q);
  float* cb  = (float*)(ws + OFF_CB);
  float* ctx = (float*)(ws + OFF_CTX);
  float* lnin = qf;  // Q f32 is dead after k_flash; reuse as LN buffer

  k_wcvt<<<(int)((8 * kMat + 255) / 256), 256, 0, stream>>>(
      Wq0, Wk0, Wv0, Wd0, Wq1, Wk1, Wv1, Wd1, wbf);
  k_gather<<<(int)((6 * kFD + 255) / 256), 256, 0, stream>>>(h, fpos, tpos, gbf);
  k_cb<<<(6 * kH * kF + 255) / 256, 256, 0, stream>>>(h, fpos, tpos, Wcb0, Wcb1, cb);

  dim3 gg(kF / 16, kD / 64);
  for (int inst = 0; inst < 6; ++inst) {
    int b = inst >> 1, dir = inst & 1;
    const bf16* gq  = gbf + (size_t)(b * 2 + dir)       * kFD;  // from-side
    const bf16* gkv = gbf + (size_t)(b * 2 + (1 - dir)) * kFD;  // to-side
    const bf16* Wqb = wbf + (size_t)(dir * 4 + 0) * kMat;
    const bf16* Wkb = wbf + (size_t)(dir * 4 + 1) * kMat;
    const bf16* Wvb = wbf + (size_t)(dir * 4 + 2) * kMat;
    const float* bv = dir ? bv1 : bv0;
    k_proj_gemm<0><<<gg, 32, 0, stream>>>(gq,  Wqb, qf + (size_t)inst * kFD, nullptr, nullptr);
    k_proj_gemm<1><<<gg, 32, 0, stream>>>(gkv, Wkb, nullptr, kbf + (size_t)inst * kFD, nullptr);
    k_proj_gemm<2><<<gg, 32, 0, stream>>>(gkv, Wvb, nullptr, vT + (size_t)inst * kFD, bv);
  }

  dim3 fg(kF / 32, kH, 6);
  k_flash<<<fg, 32, 0, stream>>>(qf, kbf, vT, cb, mix0, mix1, ctx);

  dim3 og(kF / 16, kD / 64, 6);
  k_out_proj<<<og, 32, 0, stream>>>(ctx, wbf, bd0, bd1, h, fpos, tpos, lnin);

  k_layernorm<<<dim3(kF, 6), 256, 0, stream>>>(lnin, lng0, lng1, lnb0, lnb1);
  k_mean<<<dim3(kD / 256, 6), 256, 0, stream>>>(lnin, (float*)d_out);
}
